// Transcoder_67877663146577
// MI455X (gfx1250) — matrix operand order repaired and checked by reading, not compiled, not run on hardware
//
#include <hip/hip_runtime.h>
#include <hip/hip_bf16.h>

typedef __attribute__((ext_vector_type(8)))  float  v8f;
typedef __attribute__((ext_vector_type(16))) __bf16 v16bf;
typedef __attribute__((ext_vector_type(8)))  __bf16 v8bf;
typedef __attribute__((ext_vector_type(4)))  __bf16 v4bf;

#define TM 128      // block tile M
#define TN 256      // block tile N
#define TK 32       // K step
#define LDP 40      // padded LDS row stride in bf16 elements (80B; 16B-aligned subrows)

__device__ __forceinline__ __bf16 f32_to_bf16_rne(float f) {
    unsigned u = __builtin_bit_cast(unsigned, f);
    unsigned r = u + 0x7FFFu + ((u >> 16) & 1u);   // round-to-nearest-even
    unsigned short h = (unsigned short)(r >> 16);
    return __builtin_bit_cast(__bf16, h);
}

// ---------------------------------------------------------------------------
// Kernel 1: acts = relu(x @ W^T + b) via v_wmma_f32_16x16x32_bf16
//   block tile 128x256, K-step 32, 8 waves arranged 2(M) x 4(N),
//   each wave owns a 64x64 patch = 4x4 WMMA tiles (16 WMMAs / K-step,
//   1.0 ds_load_b128 per WMMA, B-frags resident, A-frags rotating).
// ---------------------------------------------------------------------------
__global__ __launch_bounds__(256) void sae_encode_gemm(
    const float* __restrict__ x,     // [T, D]
    const float* __restrict__ W,     // [F, D]
    const float* __restrict__ b,     // [F]
    float* __restrict__ out,         // [T, F]
    int T, int D, int F)
{
    __shared__ __bf16 As[TM * LDP];     // 10 KB
    __shared__ __bf16 Bs[TN * LDP];     // 20 KB

    const int tid  = threadIdx.x;
    const int lane = tid & 31;
    const int wave = tid >> 5;
    const int wm   = wave >> 2;          // 0..1 : 64-row slab
    const int wn   = wave & 3;           // 0..3 : 64-col slab
    const int m0   = blockIdx.y * TM;
    const int n0   = blockIdx.x * TN;

    v8f acc[4][4];
#pragma unroll
    for (int mt = 0; mt < 4; ++mt)
#pragma unroll
        for (int nt = 0; nt < 4; ++nt)
            acc[mt][nt] = (v8f)(0.0f);

    // staging map: 8 threads per 32-float row -> 32 rows per slab
    const int lr = tid >> 3;             // 0..31
    const int lc = (tid & 7) * 4;        // float column 0,4,...,28

    const int hi  = lane >> 4;           // half-wave select
    const int lo4 = lane & 15;

    for (int kb = 0; kb < D; kb += TK) {
        // ---- stage A (x tile, 128x32) and B (W tile, 256x32) as bf16 ----
#pragma unroll
        for (int s = 0; s < 4; ++s) {
            const int r = lr + s * 32;
            float4 av = *(const float4*)&x[(size_t)(m0 + r) * D + kb + lc];
            v4bf ap;
            ap.x = f32_to_bf16_rne(av.x); ap.y = f32_to_bf16_rne(av.y);
            ap.z = f32_to_bf16_rne(av.z); ap.w = f32_to_bf16_rne(av.w);
            *(v4bf*)&As[r * LDP + lc] = ap;
        }
#pragma unroll
        for (int s = 0; s < 8; ++s) {
            const int r = lr + s * 32;
            float4 bv = *(const float4*)&W[(size_t)(n0 + r) * D + kb + lc];
            v4bf bp;
            bp.x = f32_to_bf16_rne(bv.x); bp.y = f32_to_bf16_rne(bv.y);
            bp.z = f32_to_bf16_rne(bv.z); bp.w = f32_to_bf16_rne(bv.w);
            *(v4bf*)&Bs[r * LDP + lc] = bp;
        }
        __syncthreads();

        // ---- B fragments (resident): 32x16 KxN bf16; lane = col n,
        //      K = hi*16 .. hi*16+15 contiguous (this kernel's order, same as A) ----
        v16bf bfrag[4];
#pragma unroll
        for (int nt = 0; nt < 4; ++nt) {
            const __bf16* p = &Bs[(wn * 64 + nt * 16 + lo4) * LDP + hi * 16];
            v8bf b0 = *(const v8bf*)p;
            v8bf b1 = *(const v8bf*)(p + 8);
            bfrag[nt] = __builtin_shufflevector(b0, b1,
                0,1,2,3,4,5,6,7,8,9,10,11,12,13,14,15);
        }

        // ---- A fragments (rotating): 16x32 MxK bf16; lane = row m,
    //      K = hi*16 .. hi*16+15 contiguous (matches B) ----
#pragma unroll
        for (int mt = 0; mt < 4; ++mt) {
            const __bf16* p = &As[(wm * 64 + mt * 16 + lo4) * LDP + hi * 16];
            v8bf a0 = *(const v8bf*)p;
            v8bf a1 = *(const v8bf*)(p + 8);
            v16bf afrag = __builtin_shufflevector(a0, a1,
                0,1,2,3,4,5,6,7,8,9,10,11,12,13,14,15);
#pragma unroll
            for (int nt = 0; nt < 4; ++nt) {
                acc[mt][nt] = __builtin_amdgcn_wmma_f32_16x16x32_bf16(
                    false, afrag, false, bfrag[nt],
                    (short)0, acc[mt][nt], false, false);
            }
        }
        __syncthreads();
    }

    // ---- epilogue: bias + ReLU; C layout: VGPR r -> (M = r + 8*hi, N = lo4) ----
#pragma unroll
    for (int nt = 0; nt < 4; ++nt) {
        const int n = n0 + wn * 64 + nt * 16 + lo4;
        const float bias = b[n];
#pragma unroll
        for (int mt = 0; mt < 4; ++mt) {
            const size_t rowbase =
                (size_t)(m0 + wm * 64 + mt * 16 + 8 * hi) * F + n;
#pragma unroll
            for (int r = 0; r < 8; ++r) {
                float v = acc[mt][nt][r] + bias;
                out[rowbase + (size_t)r * F] = v > 0.0f ? v : 0.0f;
            }
        }
    }
}

// ---------------------------------------------------------------------------
// Kernel 2: per-row top-k sparsification, in place on acts.
//   One workgroup per row; full 128KB row cached in (dynamic) LDS.
//   4-pass radix select on IEEE bits (monotonic for values >= 0),
//   index-ordered tie-breaking via per-thread-chunk exclusive scan.
// ---------------------------------------------------------------------------
__global__ __launch_bounds__(256) void topk_sparsify(
    float* __restrict__ acts, int F, const int* __restrict__ kptr)
{
    extern __shared__ float rowv[];          // F floats
    __shared__ unsigned hist[256];
    __shared__ unsigned scanbuf[256];
    __shared__ unsigned sh_prefix;
    __shared__ int sh_kk;

    const int t = threadIdx.x;
    const int k = *kptr;
    float* gro = acts + (size_t)blockIdx.x * F;

    // coalesced load row -> LDS
    for (int i = t * 4; i < F; i += blockDim.x * 4)
        *(float4*)&rowv[i] = *(const float4*)&gro[i];
    __syncthreads();

    const int chunk = F / 256;               // contiguous per-thread range
    const int base  = t * chunk;

    // ---- radix select: find bits of the k-th largest value ----
    unsigned prefix = 0;
    int kk = k;
#pragma unroll
    for (int pass = 0; pass < 4; ++pass) {
        const int shift = 24 - 8 * pass;
        const unsigned hmask = pass ? (0xFFFFFFFFu << (shift + 8)) : 0u;
        hist[t] = 0;
        __syncthreads();
        for (int j = 0; j < chunk; ++j) {
            unsigned bits = __builtin_bit_cast(unsigned, rowv[base + j]);
            if ((bits & hmask) == (prefix & hmask))
                atomicAdd(&hist[(bits >> shift) & 255u], 1u);
        }
        __syncthreads();
        if (t == 0) {
            unsigned cum = 0;
            int d = 255;
            for (; d > 0; --d) {
                unsigned c = hist[d];
                if (cum + c >= (unsigned)kk) break;
                cum += c;
            }
            sh_prefix = prefix | ((unsigned)d << shift);
            sh_kk = kk - (int)cum;           // rank within elements matching new prefix
        }
        __syncthreads();
        prefix = sh_prefix;
        kk = sh_kk;
        __syncthreads();
    }
    const unsigned thr = prefix;             // bit pattern of k-th largest

    // ---- index-ordered tie handling: exclusive scan of per-thread eq counts ----
    unsigned eq = 0;
    for (int j = 0; j < chunk; ++j) {
        unsigned bits = __builtin_bit_cast(unsigned, rowv[base + j]);
        if (bits == thr) ++eq;
    }
    scanbuf[t] = eq;
    __syncthreads();
    if (t == 0) {
        unsigned run = 0;
        for (int i = 0; i < 256; ++i) { unsigned c = scanbuf[i]; scanbuf[i] = run; run += c; }
    }
    __syncthreads();
    const unsigned tiebase = scanbuf[t];

    // ---- mask in LDS (keep > thr always; keep first kk ties by index) ----
    unsigned seen = 0;
    for (int j = 0; j < chunk; ++j) {
        float v = rowv[base + j];
        unsigned bits = __builtin_bit_cast(unsigned, v);
        bool keep;
        if (bits > thr)       keep = true;
        else if (bits == thr) { keep = (tiebase + seen) < (unsigned)kk; ++seen; }
        else                  keep = false;
        rowv[base + j] = keep ? v : 0.0f;
    }
    __syncthreads();

    // coalesced store LDS -> row
    for (int i = t * 4; i < F; i += blockDim.x * 4)
        *(float4*)&gro[i] = *(const float4*)&rowv[i];
}

// ---------------------------------------------------------------------------
extern "C" void kernel_launch(void* const* d_in, const int* in_sizes, int n_in,
                              void* d_out, int out_size, void* d_ws, size_t ws_size,
                              hipStream_t stream) {
    const float* x  = (const float*)d_in[0];
    const float* We = (const float*)d_in[1];
    const float* be = (const float*)d_in[2];
    const int*   kp = (const int*)d_in[3];
    float* out = (float*)d_out;

    const int F = in_sizes[2];
    const int D = in_sizes[1] / F;
    const int T = in_sizes[0] / D;

    dim3 grid1(F / TN, T / TM);
    sae_encode_gemm<<<grid1, 256, 0, stream>>>(x, We, be, out, T, D, F);

    const size_t lds_bytes = (size_t)F * sizeof(float);   // 128 KB row cache (<= 320 KB WGP LDS)
    topk_sparsify<<<dim3(T), 256, lds_bytes, stream>>>(out, F, kp);
}
